// de_Attention_block_20074677142010
// MI455X (gfx1250) — matrix operand order repaired and checked by reading, not compiled, not run on hardware
//
#include <hip/hip_runtime.h>

// ---------------------------------------------------------------------------
// Fused cross-attention block for MI455X (gfx1250, wave32, WMMA).
// One workgroup (256 thr = 8 waves) per batch element b in [0,2048).
// S=16, D=1024, Dp=256. All matmuls via v_wmma_f32_16x16x32_bf16.
// ---------------------------------------------------------------------------

typedef __attribute__((ext_vector_type(16))) __bf16 bf16x16;
typedef __attribute__((ext_vector_type(8)))  __bf16 bf16x8;
typedef __attribute__((ext_vector_type(8)))  float  f32x8;

#define SB   2048
#define SS   16
#define SD   1024
#define SDP  256

// LDS layout (bytes, all 16B aligned)
#define XS_STRIDE   1040            // bf16 elems per xyz row (1024 + pad)
#define PS_STRIDE   264             // bf16 elems per 256-wide row (+ pad)
#define PF_STRIDE   260             // f32 elems per pc row (+ pad)
#define OFF_XYZ     0               // [16][1040] bf16 = 33280
#define OFF_PC      33280           // [16][264]  bf16 =  8448
#define OFF_PCF     41728           // [16][260]  f32  = 16640
#define OFF_Q       58368           // [16][264]  bf16 =  8448
#define OFF_K       66816           // [16][264]  bf16 =  8448
#define OFF_X       75264           // [16][264]  bf16 =  8448
#define OFF_VT      83712           // [256][16]  bf16 =  8192
#define OFF_ATTN    91904           // [16][16]   bf16 =   512
#define SMEM_BYTES  92416

__device__ __forceinline__ bf16x16 cat8(bf16x8 lo, bf16x8 hi) {
  bf16x16 r;
#pragma unroll
  for (int i = 0; i < 8; ++i) { r[i] = lo[i]; r[i + 8] = hi[i]; }
  return r;
}

// A-fragment (16x32 bf16, M = lane&15): elems 0..15 -> K = kbase + half*16 + j
// (contiguous order, matching packed B)
__device__ __forceinline__ bf16x16 load_afrag(const __bf16* row, int kbase, int half) {
  return cat8(*(const bf16x8*)(row + kbase + half * 16),
              *(const bf16x8*)(row + kbase + half * 16 + 8));
}

// B-fragment from pre-swizzled packed weights: one 32B contiguous run per lane.
__device__ __forceinline__ bf16x16 load_bpacked(const __bf16* base, int nt, int kk,
                                                int KK, int lane) {
  const __bf16* p = base + ((size_t)((nt * KK + kk) * 32 + lane) << 4);
  return cat8(*(const bf16x8*)p, *(const bf16x8*)(p + 8));
}

__device__ __forceinline__ f32x8 wmma_bf(bf16x16 a, bf16x16 b, f32x8 c) {
  return __builtin_amdgcn_wmma_f32_16x16x32_bf16(false, a, false, b, (short)0, c,
                                                 false, false);
}

// ---------------------------------------------------------------------------
// Pre-pass: fp32 W[K][N] -> bf16 packed in per-lane WMMA-B fragment order.
// Element (k,n): kk=k/32, kr=k%32, nt=n/16, nl=n%16,
// lane = nl + (kr>=16 ? 16 : 0), j = kr%16,
// dst[((nt*KK + kk)*32 + lane)*16 + j] = bf16(W[k][n]).
// ---------------------------------------------------------------------------
__global__ void pack_weight(const float* __restrict__ W, __bf16* __restrict__ dst,
                            int K, int N) {
  int idx = blockIdx.x * blockDim.x + threadIdx.x;
  if (idx >= K * N) return;
  int k = idx / N, n = idx % N;
  int kk = k >> 5, kr = k & 31;
  int nt = n >> 4, nl = n & 15;
  int lane = nl + ((kr & 16) ? 16 : 0);
  int j = kr & 15;
  int KK = K >> 5;
  dst[(((size_t)(nt * KK + kk) * 32 + lane) << 4) + j] = (__bf16)W[idx];
}

// ---------------------------------------------------------------------------
// Main fused kernel
// ---------------------------------------------------------------------------
__global__ __launch_bounds__(256, 1)
void attn_fused(const float* __restrict__ xyz, const float* __restrict__ pc,
                const float* __restrict__ ln1g, const float* __restrict__ ln1b,
                const float* __restrict__ ln2g, const float* __restrict__ ln2b,
                const __bf16* __restrict__ wq_p, const float* __restrict__ bq,
                const __bf16* __restrict__ wk_p, const float* __restrict__ bk,
                const __bf16* __restrict__ wv_p, const float* __restrict__ bv,
                const __bf16* __restrict__ wx_p, const float* __restrict__ bx,
                float* __restrict__ out) {
  extern __shared__ char smem[];
  __bf16* s_xyz  = (__bf16*)(smem + OFF_XYZ);
  __bf16* s_pc   = (__bf16*)(smem + OFF_PC);
  float*  s_pcf  = (float*) (smem + OFF_PCF);
  __bf16* s_q    = (__bf16*)(smem + OFF_Q);
  __bf16* s_k    = (__bf16*)(smem + OFF_K);
  __bf16* s_x    = (__bf16*)(smem + OFF_X);
  __bf16* s_vT   = (__bf16*)(smem + OFF_VT);
  __bf16* s_attn = (__bf16*)(smem + OFF_ATTN);

  const int tid  = threadIdx.x;
  const int lane = tid & 31;
  const int wid  = tid >> 5;            // 0..7
  const int half = lane >> 4;           // 0 or 1
  const int l16  = lane & 15;
  const int bidx = blockIdx.x;          // batch element

  // ---------------- Phase 0: LayerNorm both streams into LDS ----------------
  // xyz: 16 rows x 1024; 2 rows per wave; 32 elems per lane per row.
#pragma unroll
  for (int rr = 0; rr < 2; ++rr) {
    int row = wid * 2 + rr;
    const float* xp = xyz + ((size_t)bidx * SS + row) * SD;
    float xs[32];
    float s = 0.f, sq = 0.f;
#pragma unroll
    for (int i = 0; i < 32; ++i) {
      float x = xp[lane + i * 32];
      xs[i] = x; s += x; sq += x * x;
    }
#pragma unroll
    for (int m = 16; m >= 1; m >>= 1) {
      s  += __shfl_xor(s,  m, 32);
      sq += __shfl_xor(sq, m, 32);
    }
    float mu   = s * (1.0f / 1024.0f);
    float rstd = rsqrtf(sq * (1.0f / 1024.0f) - mu * mu + 1e-5f);
    __bf16* orow = s_xyz + row * XS_STRIDE;
#pragma unroll
    for (int i = 0; i < 32; ++i) {
      int e = lane + i * 32;
      float y = (xs[i] - mu) * rstd * ln1g[e] + ln1b[e];
      orow[e] = (__bf16)y;
    }
  }
  // pc: 16 rows x 256; 2 rows per wave; 8 elems per lane per row.
#pragma unroll
  for (int rr = 0; rr < 2; ++rr) {
    int row = wid * 2 + rr;
    const float* pp = pc + ((size_t)bidx * SS + row) * SDP;
    float xs[8];
    float s = 0.f, sq = 0.f;
#pragma unroll
    for (int i = 0; i < 8; ++i) {
      float x = pp[lane + i * 32];
      xs[i] = x; s += x; sq += x * x;
    }
#pragma unroll
    for (int m = 16; m >= 1; m >>= 1) {
      s  += __shfl_xor(s,  m, 32);
      sq += __shfl_xor(sq, m, 32);
    }
    float mu   = s * (1.0f / 256.0f);
    float rstd = rsqrtf(sq * (1.0f / 256.0f) - mu * mu + 1e-5f);
#pragma unroll
    for (int i = 0; i < 8; ++i) {
      int e = lane + i * 32;
      float y = (xs[i] - mu) * rstd * ln2g[e] + ln2b[e];
      s_pc [row * PS_STRIDE + e] = (__bf16)y;
      s_pcf[row * PF_STRIDE + e] = y;
    }
  }
  __syncthreads();

  const int nt0 = wid;        // this wave's two N-tiles (of 16)
  const int nt1 = wid + 8;

  // ---------------- Phase A: q, k, v projections ----------------
  // q = pc_n @ Wq + bq   (K=256)
  {
    f32x8 c0 = {}, c1 = {};
    const __bf16* arow = s_pc + l16 * PS_STRIDE;
    for (int kk = 0; kk < 8; ++kk) {
      bf16x16 a = load_afrag(arow, kk * 32, half);
      c0 = wmma_bf(a, load_bpacked(wq_p, nt0, kk, 8, lane), c0);
      c1 = wmma_bf(a, load_bpacked(wq_p, nt1, kk, 8, lane), c1);
    }
#pragma unroll
    for (int t = 0; t < 2; ++t) {
      int n = (t ? nt1 : nt0) * 16 + l16;
      float bias = bq[n];
      f32x8 c = t ? c1 : c0;
#pragma unroll
      for (int r = 0; r < 8; ++r)
        s_q[(r + half * 8) * PS_STRIDE + n] = (__bf16)(c[r] + bias);
    }
  }
  // k = xyz_n @ Wk + bk   (K=1024), row-major into s_k
  {
    f32x8 c0 = {}, c1 = {};
    const __bf16* arow = s_xyz + l16 * XS_STRIDE;
    for (int kk = 0; kk < 32; ++kk) {
      bf16x16 a = load_afrag(arow, kk * 32, half);
      c0 = wmma_bf(a, load_bpacked(wk_p, nt0, kk, 32, lane), c0);
      c1 = wmma_bf(a, load_bpacked(wk_p, nt1, kk, 32, lane), c1);
    }
#pragma unroll
    for (int t = 0; t < 2; ++t) {
      int n = (t ? nt1 : nt0) * 16 + l16;
      float bias = bk[n];
      f32x8 c = t ? c1 : c0;
#pragma unroll
      for (int r = 0; r < 8; ++r)
        s_k[(r + half * 8) * PS_STRIDE + n] = (__bf16)(c[r] + bias);
    }
  }
  // v = xyz_n @ Wv + bv   (K=1024), transposed into s_vT[256][16]
  {
    f32x8 c0 = {}, c1 = {};
    const __bf16* arow = s_xyz + l16 * XS_STRIDE;
    for (int kk = 0; kk < 32; ++kk) {
      bf16x16 a = load_afrag(arow, kk * 32, half);
      c0 = wmma_bf(a, load_bpacked(wv_p, nt0, kk, 32, lane), c0);
      c1 = wmma_bf(a, load_bpacked(wv_p, nt1, kk, 32, lane), c1);
    }
#pragma unroll
    for (int t = 0; t < 2; ++t) {
      int n = (t ? nt1 : nt0) * 16 + l16;
      float bias = bv[n];
      f32x8 c = t ? c1 : c0;
      bf16x8 pk;
#pragma unroll
      for (int r = 0; r < 8; ++r) pk[r] = (__bf16)(c[r] + bias);
      *(bf16x8*)(s_vT + n * 16 + half * 8) = pk;   // vT[n][M], M = half*8 + r
    }
  }
  __syncthreads();

  // ---------------- Phase B: logits = q @ k^T / 16, softmax (wave 0) --------
  if (wid == 0) {
    f32x8 c = {};
    const __bf16* arow = s_q + l16 * PS_STRIDE;
    const __bf16* krow = s_k + l16 * PS_STRIDE;   // B: N=lane -> k row = lane
    for (int kk = 0; kk < 8; ++kk) {
      bf16x16 a = load_afrag(arow, kk * 32, half);
      bf16x16 b = cat8(*(const bf16x8*)(krow + kk * 32 + half * 16),
                       *(const bf16x8*)(krow + kk * 32 + half * 16 + 8));
      c = wmma_bf(a, b, c);
    }
    // per-row (16 lanes per half-wave) softmax
#pragma unroll
    for (int r = 0; r < 8; ++r) {
      float lv = c[r] * 0.0625f;                  // 1/sqrt(256)
      float m = lv;
#pragma unroll
      for (int msk = 8; msk >= 1; msk >>= 1) m = fmaxf(m, __shfl_xor(m, msk, 32));
      float e = __expf(lv - m);
      float sm = e;
#pragma unroll
      for (int msk = 8; msk >= 1; msk >>= 1) sm += __shfl_xor(sm, msk, 32);
      s_attn[(r + half * 8) * 16 + l16] = (__bf16)(e / sm);
    }
  }
  __syncthreads();

  // ---------------- Phase C: vals = attn @ v ; x = pc_n - vals --------------
  {
    // A = attn zero-padded to 16x32 (K=16 real, 16 zeros)
    bf16x8 alo = *(const bf16x8*)(s_attn + l16 * 16);
    bf16x8 ahi = *(const bf16x8*)(s_attn + l16 * 16 + 8);
    bf16x16 a;
#pragma unroll
    for (int i = 0; i < 8; ++i) { a[i] = half ? (__bf16)0.0f : alo[i]; a[i + 8] = half ? (__bf16)0.0f : ahi[i]; }

    f32x8 cv0 = {}, cv1 = {};
#pragma unroll
    for (int t = 0; t < 2; ++t) {
      int nt = t ? nt1 : nt0;
      bf16x16 bfr;
      if (half == 0) {                 // K=0..15 -> real v rows
        const __bf16* vp = s_vT + (nt * 16 + l16) * 16;
        bfr = cat8(*(const bf16x8*)vp, *(const bf16x8*)(vp + 8));
      } else {                         // K=16..31 -> zero padding
#pragma unroll
        for (int i = 0; i < 16; ++i) bfr[i] = (__bf16)0.0f;
      }
      if (t == 0) cv0 = wmma_bf(a, bfr, cv0);
      else        cv1 = wmma_bf(a, bfr, cv1);
    }
#pragma unroll
    for (int t = 0; t < 2; ++t) {
      int n = (t ? nt1 : nt0) * 16 + l16;
      f32x8 cv = t ? cv1 : cv0;
#pragma unroll
      for (int r = 0; r < 8; ++r) {
        int M = r + half * 8;
        float xv = s_pcf[M * PF_STRIDE + n] - cv[r];
        s_x[M * PS_STRIDE + n] = (__bf16)xv;
      }
    }
  }
  __syncthreads();

  // ---------------- Phase D: out = pc_n + (x @ Wx + bx) ---------------------
  {
    f32x8 c0 = {}, c1 = {};
    const __bf16* arow = s_x + l16 * PS_STRIDE;
    for (int kk = 0; kk < 8; ++kk) {
      bf16x16 a = load_afrag(arow, kk * 32, half);
      c0 = wmma_bf(a, load_bpacked(wx_p, nt0, kk, 8, lane), c0);
      c1 = wmma_bf(a, load_bpacked(wx_p, nt1, kk, 8, lane), c1);
    }
    float* op = out + (size_t)bidx * (SS * SDP);
#pragma unroll
    for (int t = 0; t < 2; ++t) {
      int n = (t ? nt1 : nt0) * 16 + l16;
      float bias = bx[n];
      f32x8 c = t ? c1 : c0;
#pragma unroll
      for (int r = 0; r < 8; ++r) {
        int M = r + half * 8;
        op[M * SDP + n] = s_pcf[M * PF_STRIDE + n] + c[r] + bias;
      }
    }
  }
}

// ---------------------------------------------------------------------------
// Host launcher
// ---------------------------------------------------------------------------
extern "C" void kernel_launch(void* const* d_in, const int* in_sizes, int n_in,
                              void* d_out, int out_size, void* d_ws, size_t ws_size,
                              hipStream_t stream) {
  const float* xyz  = (const float*)d_in[0];
  const float* pc   = (const float*)d_in[1];
  const float* ln1g = (const float*)d_in[2];
  const float* ln1b = (const float*)d_in[3];
  const float* ln2g = (const float*)d_in[4];
  const float* ln2b = (const float*)d_in[5];
  const float* Wq   = (const float*)d_in[6];
  const float* bq   = (const float*)d_in[7];
  const float* Wk   = (const float*)d_in[8];
  const float* bk   = (const float*)d_in[9];
  const float* Wv   = (const float*)d_in[10];
  const float* bv   = (const float*)d_in[11];
  const float* Wx   = (const float*)d_in[12];
  const float* bx   = (const float*)d_in[13];
  float* out = (float*)d_out;

  __bf16* ws   = (__bf16*)d_ws;
  __bf16* wq_p = ws;                       //  65536 elems
  __bf16* wk_p = ws + 65536;               // 262144 elems
  __bf16* wv_p = ws + 65536 + 262144;      // 262144 elems
  __bf16* wx_p = ws + 65536 + 2 * 262144;  //  65536 elems

  hipFuncSetAttribute((const void*)attn_fused,
                      hipFuncAttributeMaxDynamicSharedMemorySize, SMEM_BYTES);

  pack_weight<<<(256 * 256 + 255) / 256, 256, 0, stream>>>(Wq, wq_p, 256, 256);
  pack_weight<<<(1024 * 256 + 255) / 256, 256, 0, stream>>>(Wk, wk_p, 1024, 256);
  pack_weight<<<(1024 * 256 + 255) / 256, 256, 0, stream>>>(Wv, wv_p, 1024, 256);
  pack_weight<<<(256 * 256 + 255) / 256, 256, 0, stream>>>(Wx, wx_p, 256, 256);

  attn_fused<<<SB, 256, SMEM_BYTES, stream>>>(xyz, pc, ln1g, ln1b, ln2g, ln2b,
                                              wq_p, bq, wk_p, bk, wv_p, bv,
                                              wx_p, bx, out);
}
